// minLSTM_32040456028359
// MI455X (gfx1250) — compile-verified, matrix operand order checked by reading, not run on hardware
//
#include <hip/hip_runtime.h>

// ---------------------------------------------------------------------------
// minLSTM forward for MI455X (gfx1250, wave32, WMMA, async-to-LDS)
//   B=4, S=4096, D=1024, H=2048, M=B*S=16384
// Pipeline:
//   1) f32 -> bf16 conversion of x, W_fwd, W_out (workspace)
//   2) GEMM1 (WMMA bf16, LDS double-buffered, async copies): fused epilogue
//      computes log_f / log_values (never materializes 384MB gates tensor)
//   3) 3-pass chunked log-space scan over SEQ (8 chunks of 512)
//   4) GEMM2 (WMMA bf16): out = h @ W_out^T, f32 output
// ---------------------------------------------------------------------------

typedef __bf16 bf16_t;
typedef __attribute__((ext_vector_type(16))) __bf16 v16bf;
typedef __attribute__((ext_vector_type(8)))  __bf16 v8bf;
typedef __attribute__((ext_vector_type(8)))  float  v8f;

union FragBF { v8bf h[2]; v16bf v; };

#define MTOT   16384
#define DIN    1024
#define HID    2048
#define SEQ    4096
#define BATCH  4
#define NCH    8           // scan chunks
#define TCH    512         // SEQ / NCH

#define KB     32          // K per pipeline stage (one WMMA K-step)
#define PITCH  40          // 32 bf16 + 8 pad: 20-DWORD lane stride -> conflict-free
                           // b128 LDS reads, 16B-aligned chunk addresses
#define BSTRIP (64 * PITCH) // elements per 64-column B strip buffer

// --------------------------- async-to-LDS plumbing -------------------------
#if defined(__has_builtin)
# if __has_builtin(__builtin_amdgcn_global_load_async_to_lds_b128) && \
     __has_builtin(__builtin_amdgcn_s_wait_asynccnt)
#  define USE_ASYNC 1
# endif
#endif
#ifndef USE_ASYNC
# define USE_ASYNC 0
#endif

#if USE_ASYNC
typedef int v4i_t __attribute__((vector_size(16)));
typedef __attribute__((address_space(1))) v4i_t* g128_t;   // global src
typedef __attribute__((address_space(3))) v4i_t* l128_t;   // LDS dst
# define WAIT_ASYNC(n) __builtin_amdgcn_s_wait_asynccnt(n)
#else
# define WAIT_ASYNC(n) ((void)0)
#endif

__device__ __forceinline__ void cp16(const bf16_t* g, bf16_t* l) {
#if USE_ASYNC
    // generic->AS casts via integers: AS1 keeps full 64-bit VA; AS3 is the
    // low-32-bit LDS offset of the generic pointer.
    __builtin_amdgcn_global_load_async_to_lds_b128(
        (g128_t)(unsigned long long)g,
        (l128_t)(unsigned int)(unsigned long long)l, 0, 0);
#else
    *(v8bf*)l = *(const v8bf*)g;
#endif
}

// ---------------------------------------------------------------------------
// f32 -> bf16, 8 elements per thread (b128 in / b128 out)
// ---------------------------------------------------------------------------
__global__ void __launch_bounds__(256)
cvt_f32_to_bf16(const float* __restrict__ in, bf16_t* __restrict__ out, int n) {
    int i = (blockIdx.x * blockDim.x + threadIdx.x) * 8;
    if (i + 8 <= n) {
        float4 a = *(const float4*)(in + i);
        float4 b = *(const float4*)(in + i + 4);
        v8bf o;
        o[0] = (bf16_t)a.x; o[1] = (bf16_t)a.y; o[2] = (bf16_t)a.z; o[3] = (bf16_t)a.w;
        o[4] = (bf16_t)b.x; o[5] = (bf16_t)b.y; o[6] = (bf16_t)b.z; o[7] = (bf16_t)b.w;
        *(v8bf*)(out + i) = o;
    }
}

// --------------------------- numerics helpers ------------------------------
__device__ __forceinline__ float softplusf(float x) {
    return fmaxf(x, 0.f) + log1pf(__expf(-fabsf(x)));
}
__device__ __forceinline__ float logaddexpf_(float a, float b) {
    float mx = fmaxf(a, b);
    float mn = fminf(a, b);
    float r  = mx + log1pf(__expf(mn - mx));
    return (mn == -INFINITY) ? mx : r;
}

// --------------------------- WMMA fragment helpers -------------------------
// A (16x32 bf16, MxK): lanes 0-15 hold M=lane, K{0..7,16..23};
//                      lanes 16-31 hold M=lane-16, K{8..15,24..31}
// B (32x16 bf16, KxN): mirrored striping; B column n is contiguous along K in
//                      LDS (column = weight-matrix row)
__device__ __forceinline__ v16bf frag_lds(const bf16_t* base, int idx0, int l15, int kh) {
    FragBF f;
    const bf16_t* p = base + (idx0 + l15) * PITCH + kh * 8;
    f.h[0] = *(const v8bf*)(p);
    f.h[1] = *(const v8bf*)(p + 16);
    return f.v;
}
__device__ __forceinline__ v8f wmma_bf16(v16bf a, v16bf b, v8f c) {
    return __builtin_amdgcn_wmma_f32_16x16x32_bf16(false, a, false, b, (short)0, c,
                                                   false, false);
}

// ---------------------------------------------------------------------------
// GEMM1 + fused gate transform.
// grid = (M/128, HID/64), block = 256 (8 waves = 4 M-waves x 2 N-waves).
// Block tile: 128 M x 64 N, x3 gate strips (f @ col, i @ col+H, hid @ col+2H);
// wave tile 32x32 per strip -> 12 accumulators. A and B slabs double-buffered
// in LDS, filled with async-to-LDS copies (5 b128 per thread per stage).
// ---------------------------------------------------------------------------
__global__ void __launch_bounds__(256)
gemm_gates_fused(const bf16_t* __restrict__ Xb, const bf16_t* __restrict__ Wf,
                 float* __restrict__ log_f_out, float* __restrict__ log_v_out) {
    __shared__ bf16_t sA[2][128 * PITCH];     // 2 x 10240 B
    __shared__ bf16_t sB[2][3 * BSTRIP];      // 2 x 15360 B  (total ~50 KB)

    const int m0   = blockIdx.x * 128;
    const int n0   = blockIdx.y * 64;
    const int tid  = threadIdx.x;
    const int lane = tid & 31;
    const int wave = tid >> 5;
    const int wm   = wave >> 1;               // 0..3  (M sub-block)
    const int wn   = wave & 1;                // 0..1  (N sub-block)
    const int l15  = lane & 15;
    const int kh   = lane >> 4;

    v8f z = {0.f,0.f,0.f,0.f,0.f,0.f,0.f,0.f};
    v8f acc[3][2][2];
    #pragma unroll
    for (int g = 0; g < 3; ++g)
        #pragma unroll
        for (int mt = 0; mt < 2; ++mt)
            #pragma unroll
            for (int nt = 0; nt < 2; ++nt) acc[g][mt][nt] = z;

    auto issue = [&](int kb, int bi) {
        // A slab: 128 rows x 4 chunks = 512 b128 -> 2 per thread
        #pragma unroll
        for (int j = 0; j < 2; ++j) {
            int id  = tid + 256 * j;
            int row = id >> 2, ck = id & 3;
            cp16(Xb + (size_t)(m0 + row) * DIN + kb + ck * 8,
                 &sA[bi][row * PITCH + ck * 8]);
        }
        // B slabs: 3 strips x 64 cols x 4 chunks = 768 b128 -> 3 per thread
        #pragma unroll
        for (int j = 0; j < 3; ++j) {
            int id  = tid + 256 * j;
            int g   = id >> 8;                // strip
            int rem = id & 255;
            int col = rem >> 2, ck = rem & 3;
            cp16(Wf + (size_t)(g * HID + n0 + col) * DIN + kb + ck * 8,
                 &sB[bi][g * BSTRIP + col * PITCH + ck * 8]);
        }
    };

    const int NST = DIN / KB;                 // 32 stages
    issue(0, 0);
    for (int s = 0; s < NST; ++s) {
        const int bi = s & 1;
        if (s + 1 < NST) {
            issue((s + 1) * KB, bi ^ 1);
            WAIT_ASYNC(5);                    // stage s landed; stage s+1 in flight
        } else {
            WAIT_ASYNC(0);
        }
        __syncthreads();

        v16bf a0 = frag_lds(&sA[bi][0], wm * 32 + 0,  l15, kh);
        v16bf a1 = frag_lds(&sA[bi][0], wm * 32 + 16, l15, kh);
        #pragma unroll
        for (int nt = 0; nt < 2; ++nt) {
            #pragma unroll
            for (int g = 0; g < 3; ++g) {
                v16bf b = frag_lds(&sB[bi][g * BSTRIP], wn * 32 + nt * 16, l15, kh);
                acc[g][0][nt] = wmma_bf16(a0, b, acc[g][0][nt]);
                acc[g][1][nt] = wmma_bf16(a1, b, acc[g][1][nt]);
            }
        }
        __syncthreads();                      // protect buffer re-fill
    }

    // epilogue: C/D layout -> VGPR r, lanes 0-15: M=r, lanes 16-31: M=r+8
    #pragma unroll
    for (int mt = 0; mt < 2; ++mt) {
        #pragma unroll
        for (int nt = 0; nt < 2; ++nt) {
            const int col = n0 + wn * 32 + nt * 16 + l15;
            #pragma unroll
            for (int r = 0; r < 8; ++r) {
                const int gm = m0 + wm * 32 + mt * 16 + r + 8 * kh;
                float fg = acc[0][mt][nt][r];
                float ig = acc[1][mt][nt][r];
                float hg = acc[2][mt][nt][r];
                float diff = softplusf(-fg) - softplusf(-ig);
                float lf   = -softplusf(diff);
                float li   = -softplusf(-diff);
                float lg   = (hg >= 0.f) ? __logf(hg + 0.5f) : -softplusf(-hg);
                size_t o = (size_t)gm * HID + col;
                log_f_out[o] = lf;
                log_v_out[o] = li + lg;
            }
        }
    }
}

// ---------------------------------------------------------------------------
// Chunked log-space scan, 3 passes.
//   pass1: per (b, chunk, h): A = sum(log_f), B = local scan with -inf init
//   pass2: per (b, h): serial scan over 8 chunk summaries -> carry-in
//   pass3: per (b, chunk, h): replay local scan seeded with carry-in, emit h
// ---------------------------------------------------------------------------
__global__ void __launch_bounds__(256)
scan_pass1(const float* __restrict__ log_f, const float* __restrict__ log_v,
           float* __restrict__ Asum, float* __restrict__ Bval) {
    const int idx = blockIdx.x * blockDim.x + threadIdx.x;   // BATCH*NCH*HID
    const int h = idx & (HID - 1);
    const int c = (idx >> 11) & (NCH - 1);
    const int b = idx >> 14;
    const size_t base = ((size_t)b * SEQ + (size_t)c * TCH) * HID + h;
    float A = 0.f, Bv = -INFINITY;
    for (int t = 0; t < TCH; ++t) {
        const size_t o = base + (size_t)t * HID;
        float f = log_f[o], v = log_v[o];
        A += f;
        Bv = logaddexpf_(Bv + f, v);
    }
    Asum[idx] = A;
    Bval[idx] = Bv;
}

__global__ void __launch_bounds__(256)
scan_pass2(const float* __restrict__ Asum, const float* __restrict__ Bval,
           float* __restrict__ carry) {
    const int idx = blockIdx.x * blockDim.x + threadIdx.x;   // BATCH*HID
    const int b = idx >> 11;
    const int h = idx & (HID - 1);
    float st = -INFINITY;
    #pragma unroll
    for (int c = 0; c < NCH; ++c) {
        const size_t o = ((size_t)(b * NCH + c)) * HID + h;
        carry[o] = st;
        st = logaddexpf_(st + Asum[o], Bval[o]);
    }
}

__global__ void __launch_bounds__(256)
scan_pass3(const float* __restrict__ log_f, const float* __restrict__ log_v,
           const float* __restrict__ carry, bf16_t* __restrict__ h_out) {
    const int idx = blockIdx.x * blockDim.x + threadIdx.x;   // BATCH*NCH*HID
    const int h = idx & (HID - 1);
    const int c = (idx >> 11) & (NCH - 1);
    const int b = idx >> 14;
    const size_t base = ((size_t)b * SEQ + (size_t)c * TCH) * HID + h;
    float logh = carry[idx];
    for (int t = 0; t < TCH; ++t) {
        const size_t o = base + (size_t)t * HID;
        logh = logaddexpf_(logh + log_f[o], log_v[o]);
        h_out[o] = (bf16_t)__expf(logh);
    }
}

// ---------------------------------------------------------------------------
// GEMM2: out = h @ W_out^T.  grid = (M/128, DIN/64), same pipeline as GEMM1
// with a single strip (3 b128 async copies per thread per stage).
// ---------------------------------------------------------------------------
__global__ void __launch_bounds__(256)
gemm_out(const bf16_t* __restrict__ Hb, const bf16_t* __restrict__ Wo,
         float* __restrict__ out) {
    __shared__ bf16_t sA[2][128 * PITCH];
    __shared__ bf16_t sB[2][BSTRIP];

    const int m0   = blockIdx.x * 128;
    const int n0   = blockIdx.y * 64;
    const int tid  = threadIdx.x;
    const int lane = tid & 31;
    const int wave = tid >> 5;
    const int wm   = wave >> 1;
    const int wn   = wave & 1;
    const int l15  = lane & 15;
    const int kh   = lane >> 4;

    v8f z = {0.f,0.f,0.f,0.f,0.f,0.f,0.f,0.f};
    v8f acc[2][2];
    #pragma unroll
    for (int mt = 0; mt < 2; ++mt)
        #pragma unroll
        for (int nt = 0; nt < 2; ++nt) acc[mt][nt] = z;

    auto issue = [&](int kb, int bi) {
        #pragma unroll
        for (int j = 0; j < 2; ++j) {
            int id  = tid + 256 * j;
            int row = id >> 2, ck = id & 3;
            cp16(Hb + (size_t)(m0 + row) * HID + kb + ck * 8,
                 &sA[bi][row * PITCH + ck * 8]);
        }
        {
            int col = tid >> 2, ck = tid & 3;   // 64 cols x 4 chunks = 256
            cp16(Wo + (size_t)(n0 + col) * HID + kb + ck * 8,
                 &sB[bi][col * PITCH + ck * 8]);
        }
    };

    const int NST = HID / KB;                 // 64 stages
    issue(0, 0);
    for (int s = 0; s < NST; ++s) {
        const int bi = s & 1;
        if (s + 1 < NST) {
            issue((s + 1) * KB, bi ^ 1);
            WAIT_ASYNC(3);
        } else {
            WAIT_ASYNC(0);
        }
        __syncthreads();

        v16bf a0 = frag_lds(&sA[bi][0], wm * 32 + 0,  l15, kh);
        v16bf a1 = frag_lds(&sA[bi][0], wm * 32 + 16, l15, kh);
        #pragma unroll
        for (int nt = 0; nt < 2; ++nt) {
            v16bf b = frag_lds(&sB[bi][0], wn * 32 + nt * 16, l15, kh);
            acc[0][nt] = wmma_bf16(a0, b, acc[0][nt]);
            acc[1][nt] = wmma_bf16(a1, b, acc[1][nt]);
        }
        __syncthreads();
    }

    #pragma unroll
    for (int mt = 0; mt < 2; ++mt) {
        #pragma unroll
        for (int nt = 0; nt < 2; ++nt) {
            const int col = n0 + wn * 32 + nt * 16 + l15;
            #pragma unroll
            for (int r = 0; r < 8; ++r) {
                const int gm = m0 + wm * 32 + mt * 16 + r + 8 * kh;
                out[(size_t)gm * DIN + col] = acc[mt][nt][r];
            }
        }
    }
}

// ---------------------------------------------------------------------------
// launch
// ---------------------------------------------------------------------------
extern "C" void kernel_launch(void* const* d_in, const int* in_sizes, int n_in,
                              void* d_out, int out_size, void* d_ws, size_t ws_size,
                              hipStream_t stream) {
    const float* x  = (const float*)d_in[0];   // [4,4096,1024]
    const float* Wf = (const float*)d_in[1];   // [6144,1024]
    const float* Wo = (const float*)d_in[2];   // [1024,2048]
    float* out = (float*)d_out;                // [4,4096,1024]

    char* ws = (char*)d_ws;
    bf16_t* xb    = (bf16_t*)(ws + 0);           //  32 MB
    bf16_t* wf    = (bf16_t*)(ws + 33554432);    //  12 MB
    bf16_t* wo    = (bf16_t*)(ws + 46137344);    //   4 MB
    float*  lf    = (float*) (ws + 50331648);    // 128 MB
    float*  lv    = (float*) (ws + 184549376);   // 128 MB
    bf16_t* hb    = (bf16_t*)(ws + 318767104);   //  64 MB
    float*  scA   = (float*) (ws + 385875968);   // 256 KB chunk A-sums
    float*  scB   = (float*) (ws + 386138112);   // 256 KB chunk B-vals
    float*  carry = (float*) (ws + 386400256);   // 256 KB carry-in

    const int nX  = MTOT * DIN;
    const int nWf = 3 * HID * DIN;
    const int nWo = DIN * HID;

    cvt_f32_to_bf16<<<nX  / 2048, 256, 0, stream>>>(x,  xb, nX);
    cvt_f32_to_bf16<<<nWf / 2048, 256, 0, stream>>>(Wf, wf, nWf);
    cvt_f32_to_bf16<<<nWo / 2048, 256, 0, stream>>>(Wo, wo, nWo);

    gemm_gates_fused<<<dim3(MTOT / 128, HID / 64), 256, 0, stream>>>(xb, wf, lf, lv);

    scan_pass1<<<(BATCH * NCH * HID) / 256, 256, 0, stream>>>(lf, lv, scA, scB);
    scan_pass2<<<(BATCH * HID) / 256, 256, 0, stream>>>(scA, scB, carry);
    scan_pass3<<<(BATCH * NCH * HID) / 256, 256, 0, stream>>>(lf, lv, carry, hb);

    gemm_out<<<dim3(MTOT / 128, DIN / 64), 256, 0, stream>>>(hb, wo, out);
}
